// Ourmethod_32908039422210
// MI455X (gfx1250) — matrix operand order repaired and checked by reading, not compiled, not run on hardware
//
#include <hip/hip_runtime.h>
#include <hip/hip_bf16.h>
#include <math.h>

// ---------------------------------------------------------------------------
// MeshNet-like forward pass for MI455X (gfx1250, wave32, WMMA).
//
// Heavy linear algebra (adjacency matmuls ~360 GFLOP, KNN Gram matrix,
// EdgeConv 1x1 convs) uses V_WMMA_F32_16X16X32_BF16 with f32 accumulation.
// Adjacency (binary, exact in bf16) is 128 MB -> resident in the 192 MB L2.
// All WMMA RHS operands are stored transposed (M x K, bf16) so operand loads
// are contiguous 16B chunks in the same k order as the A loads.
// The GEMM K-loop is branch-free: EXEC stays all-ones across every WMMA
// (ISA requirement) and accumulators never spill.
// Requires ws_size >= ~640 MB.
// ---------------------------------------------------------------------------

typedef __bf16 bf16t;
typedef __attribute__((ext_vector_type(16))) __bf16 v16bf;
typedef __attribute__((ext_vector_type(8)))  __bf16 v8bf;
typedef __attribute__((ext_vector_type(8)))  float  v8f;

#define TPB 256
static const int NPTS = 8000;

static __device__ __forceinline__ v16bf cat8(v8bf a, v8bf b) {
  return __builtin_shufflevector(a, b, 0,1,2,3,4,5,6,7,8,9,10,11,12,13,14,15);
}

// ---------------------------------------------------------------------------
// WMMA GEMM:  C[O x M] = A[O x K] * B^T  (B stored as BT[M x K], bf16)
// Each wave computes a 16 x 64 strip (4 16x16 tiles, A reused across tiles).
// Block = 128 threads = 4 waves.
// Out-of-range column tiles are CLAMPED to column 0 (B reads stay in valid
// memory, results discarded at store time) so the K-loop has no control flow:
// no exec masking around WMMA, no accumulator spills.
// A operand, this code's k order (not AMD's table order): row = lane%16; contiguous 16 elems at 16*kh.
// B operand, same k order as A: col = lane%16; contiguous 16 elems at 16*kh.
// D layout: element i -> row 8*kh + i, col lane%16.
// ---------------------------------------------------------------------------
__global__ __launch_bounds__(128)
void wmma_mm_kernel(const bf16t* __restrict__ A, const bf16t* __restrict__ B,
                    float* __restrict__ C, const float* __restrict__ bias,
                    int K, int M) {
  const int lane = threadIdx.x & 31;
  const int wave = threadIdx.x >> 5;
  const int rowbase = blockIdx.y * 16;
  const int colbase = blockIdx.x * 256 + wave * 64;
  const int r  = lane & 15;
  const int kh = lane >> 4;

  const int cb0 = colbase;
  const int cb1 = colbase + 16;
  const int cb2 = colbase + 32;
  const int cb3 = colbase + 48;

  const bf16t* ap  = A + (size_t)(rowbase + r) * K + kh * 16;
  const bf16t* bp0 = B + (size_t)((cb0 < M ? cb0 : 0) + r) * K + kh * 16;
  const bf16t* bp1 = B + (size_t)((cb1 < M ? cb1 : 0) + r) * K + kh * 16;
  const bf16t* bp2 = B + (size_t)((cb2 < M ? cb2 : 0) + r) * K + kh * 16;
  const bf16t* bp3 = B + (size_t)((cb3 < M ? cb3 : 0) + r) * K + kh * 16;

  v8f acc0 = {}, acc1 = {}, acc2 = {}, acc3 = {};

  for (int k0 = 0; k0 < K; k0 += 32) {
    v16bf a = cat8(*(const v8bf*)(ap), *(const v8bf*)(ap + 8));
    ap += 32;
    v16bf b0 = cat8(*(const v8bf*)(bp0), *(const v8bf*)(bp0 + 8));
    bp0 += 32;
    acc0 = __builtin_amdgcn_wmma_f32_16x16x32_bf16(false, a, false, b0, (short)0, acc0, false, false);
    v16bf b1 = cat8(*(const v8bf*)(bp1), *(const v8bf*)(bp1 + 8));
    bp1 += 32;
    acc1 = __builtin_amdgcn_wmma_f32_16x16x32_bf16(false, a, false, b1, (short)0, acc1, false, false);
    v16bf b2 = cat8(*(const v8bf*)(bp2), *(const v8bf*)(bp2 + 8));
    bp2 += 32;
    acc2 = __builtin_amdgcn_wmma_f32_16x16x32_bf16(false, a, false, b2, (short)0, acc2, false, false);
    v16bf b3 = cat8(*(const v8bf*)(bp3), *(const v8bf*)(bp3 + 8));
    bp3 += 32;
    acc3 = __builtin_amdgcn_wmma_f32_16x16x32_bf16(false, a, false, b3, (short)0, acc3, false, false);
  }

  // Store epilogue: guards here are wave-uniform and occur after all WMMAs.
  float bv[8];
#pragma unroll
  for (int i = 0; i < 8; ++i) bv[i] = bias ? bias[rowbase + kh * 8 + i] : 0.f;

  if (cb0 < M) {
#pragma unroll
    for (int i = 0; i < 8; ++i)
      C[(size_t)(rowbase + kh * 8 + i) * M + cb0 + r] = acc0[i] + bv[i];
  }
  if (cb1 < M) {
#pragma unroll
    for (int i = 0; i < 8; ++i)
      C[(size_t)(rowbase + kh * 8 + i) * M + cb1 + r] = acc1[i] + bv[i];
  }
  if (cb2 < M) {
#pragma unroll
    for (int i = 0; i < 8; ++i)
      C[(size_t)(rowbase + kh * 8 + i) * M + cb2 + r] = acc2[i] + bv[i];
  }
  if (cb3 < M) {
#pragma unroll
    for (int i = 0; i < 8; ++i)
      C[(size_t)(rowbase + kh * 8 + i) * M + cb3 + r] = acc3[i] + bv[i];
  }
}

// Scalar GEMM fallback for tiny dims (K=12 stems, O in {1,8,32,128@M=1}).
// act: 0=none 1=lrelu 2=sigmoid 3=relu
__global__ void smm_kernel(const float* __restrict__ W, const float* __restrict__ X,
                           const float* __restrict__ bias, float* __restrict__ out,
                           int O, int K, int M, int act) {
  long i = (long)blockIdx.x * TPB + threadIdx.x;
  if (i >= (long)O * M) return;
  int o = (int)(i / M), m = (int)(i % M);
  float s = bias ? bias[o] : 0.f;
  for (int k = 0; k < K; ++k) s += W[o * K + k] * X[(size_t)k * M + m];
  if (act == 1) s = s > 0.f ? s : 0.2f * s;
  else if (act == 2) s = 1.f / (1.f + expf(-s));
  else if (act == 3) s = s > 0.f ? s : 0.f;
  out[i] = s;
}

// Per-channel mean/var (biased) over M spatial elements.
__global__ void stats_kernel(const float* __restrict__ Y, float* __restrict__ mean,
                             float* __restrict__ var, int M) {
  __shared__ float ss[TPB], ss2[TPB];
  int c = blockIdx.x;
  const float* row = Y + (size_t)c * M;
  float s = 0.f, s2 = 0.f;
  for (int i = threadIdx.x; i < M; i += TPB) { float v = row[i]; s += v; s2 += v * v; }
  ss[threadIdx.x] = s; ss2[threadIdx.x] = s2;
  __syncthreads();
  for (int st = TPB / 2; st > 0; st >>= 1) {
    if (threadIdx.x < st) { ss[threadIdx.x] += ss[threadIdx.x + st]; ss2[threadIdx.x] += ss2[threadIdx.x + st]; }
    __syncthreads();
  }
  if (threadIdx.x == 0) { float m = ss[0] / M; mean[c] = m; var[c] = ss2[0] / M - m * m; }
}

// BN + leaky-relu (+ optional residual added after activation). In-place safe.
__global__ void bn_act_kernel(const float* __restrict__ Y, const float* __restrict__ g,
                              const float* __restrict__ b, const float* __restrict__ mean,
                              const float* __restrict__ var, const float* __restrict__ resid,
                              float* __restrict__ out, int M, long total) {
  long i = (long)blockIdx.x * TPB + threadIdx.x;
  if (i >= total) return;
  int c = (int)(i / M);
  float v = (Y[i] - mean[c]) * rsqrtf(var[c] + 1e-5f) * g[c] + b[c];
  v = v > 0.f ? v : 0.2f * v;
  if (resid) v += resid[i];
  out[i] = v;
}

__global__ void f32_to_bf16_kernel(const float* __restrict__ s, bf16t* __restrict__ d, long n) {
  long i = (long)blockIdx.x * TPB + threadIdx.x;
  if (i < n) d[i] = (bf16t)s[i];
}

// (O x M) f32  ->  (M x O) bf16
__global__ void tr_bf16_kernel(const float* __restrict__ s, bf16t* __restrict__ d, int O, int M) {
  long i = (long)blockIdx.x * TPB + threadIdx.x;
  if (i >= (long)O * M) return;
  int o = (int)(i / M), m = (int)(i % M);
  d[(size_t)m * O + o] = (bf16t)s[i];
}

// xx[n] = sum_c x[c][n]^2
__global__ void colsumsq_kernel(const float* __restrict__ x, float* __restrict__ xx, int C, int N) {
  int n = blockIdx.x * TPB + threadIdx.x;
  if (n >= N) return;
  float s = 0.f;
  for (int c = 0; c < C; ++c) { float v = x[(size_t)c * N + n]; s += v * v; }
  xx[n] = s;
}

// pd = 2*inner - xx[n] - xx[m]   (in place over the Gram matrix)
__global__ void pd_kernel(float* __restrict__ pd, const float* __restrict__ xx, int N) {
  long i = (long)blockIdx.x * TPB + threadIdx.x;
  if (i >= (long)N * N) return;
  int n = (int)(i / N), m = (int)(i % N);
  pd[i] = 2.f * pd[i] - xx[n] - xx[m];
}

// Row-wise top-k (descending, tie -> lowest index) via iterative argmax in LDS.
__global__ void topk_kernel(const float* __restrict__ pd, int* __restrict__ idx, int N, int K) {
  __shared__ float sv[8000];
  __shared__ float rv[TPB];
  __shared__ int   ri[TPB];
  int n = blockIdx.x;
  const float* row = pd + (size_t)n * N;
  for (int i = threadIdx.x; i < N; i += TPB) sv[i] = row[i];
  __syncthreads();
  for (int t = 0; t < K; ++t) {
    float bv = -INFINITY; int bi = 0x7fffffff;
    for (int i = threadIdx.x; i < N; i += TPB) {
      float v = sv[i];
      if (v > bv) { bv = v; bi = i; }
    }
    rv[threadIdx.x] = bv; ri[threadIdx.x] = bi;
    __syncthreads();
    for (int s = TPB / 2; s > 0; s >>= 1) {
      if (threadIdx.x < s) {
        float ov = rv[threadIdx.x + s]; int oi = ri[threadIdx.x + s];
        if (ov > rv[threadIdx.x] || (ov == rv[threadIdx.x] && oi < ri[threadIdx.x])) {
          rv[threadIdx.x] = ov; ri[threadIdx.x] = oi;
        }
      }
      __syncthreads();
    }
    if (threadIdx.x == 0) { idx[(size_t)n * K + t] = ri[0]; sv[ri[0]] = -INFINITY; }
    __syncthreads();
  }
}

// adj[n][m] = any of 9 face-index equalities (symmetric) -> bf16 {0,1}
__global__ void adj_gen_kernel(const int* __restrict__ face, bf16t* __restrict__ adj, int N) {
  long i = (long)blockIdx.x * TPB + threadIdx.x;
  if (i >= (long)N * N) return;
  int n = (int)(i / N), m = (int)(i % N);
  int a0 = face[n * 3], a1 = face[n * 3 + 1], a2 = face[n * 3 + 2];
  int b0 = face[m * 3], b1 = face[m * 3 + 1], b2 = face[m * 3 + 2];
  bool e = a0 == b0 || a0 == b1 || a0 == b2 || a1 == b0 || a1 == b1 || a1 == b2 ||
           a2 == b0 || a2 == b1 || a2 == b2;
  adj[i] = e ? (bf16t)1.0f : (bf16t)0.0f;
}

// EdgeConv feature: featT[(n*K+k)][c] = (c<64) x[c][idx]-x[c][n] else x[c-64][n]
__global__ void gather_feat_kernel(const float* __restrict__ x, const int* __restrict__ idx,
                                   bf16t* __restrict__ featT, int N, int K) {
  long i = (long)blockIdx.x * TPB + threadIdx.x;
  long total = (long)N * K * 128;
  if (i >= total) return;
  int c = (int)(i & 127);
  long m = i >> 7;
  int k = (int)(m % K), n = (int)(m / K);
  int j = idx[(size_t)n * K + k];
  float v;
  if (c < 64) v = x[(size_t)c * N + j] - x[(size_t)c * N + n];
  else        v = x[(size_t)(c - 64) * N + n];
  featT[i] = (bf16t)v;
}

// m5 input: concat(f2[64ch per (n,k)], broadcast xk1[64ch per n]) -> (N*K) x 128 bf16
__global__ void build_cat_kernel(const float* __restrict__ f2, const float* __restrict__ xk1,
                                 bf16t* __restrict__ outT, int N, int K) {
  long i = (long)blockIdx.x * TPB + threadIdx.x;
  long total = (long)N * K * 128;
  if (i >= total) return;
  int c = (int)(i & 127);
  long m = i >> 7;
  int n = (int)(m / K);
  float v = (c < 64) ? f2[(size_t)c * (N * K) + m] : xk1[(size_t)(c - 64) * N + n];
  outT[i] = (bf16t)v;
}

// out[c][n] = max_k f[c][n*K+k]
__global__ void maxk_kernel(const float* __restrict__ f, float* __restrict__ out,
                            int C, int N, int K) {
  long i = (long)blockIdx.x * TPB + threadIdx.x;
  if (i >= (long)C * N) return;
  int c = (int)(i / N), n = (int)(i % N);
  const float* p = f + (size_t)c * N * K + (size_t)n * K;
  float mx = p[0];
  for (int k = 1; k < K; ++k) mx = fmaxf(mx, p[k]);
  out[i] = mx;
}

__global__ void combine_kernel(const float* __restrict__ coor, const float* __restrict__ x1,
                               const float* __restrict__ attn, const float* __restrict__ attc,
                               float* __restrict__ f, int N) {
  long i = (long)blockIdx.x * TPB + threadIdx.x;
  if (i >= (long)128 * N) return;
  int n = (int)(i % N);
  float c = coor[i], x = x1[i];
  f[i] = c + c * attn[n] + x + x * attc[n];
}

__global__ void meann_kernel(const float* __restrict__ f, float* __restrict__ y, int M) {
  __shared__ float s[TPB];
  int c = blockIdx.x;
  float acc = 0.f;
  for (int i = threadIdx.x; i < M; i += TPB) acc += f[(size_t)c * M + i];
  s[threadIdx.x] = acc; __syncthreads();
  for (int st = TPB / 2; st > 0; st >>= 1) {
    if (threadIdx.x < st) s[threadIdx.x] += s[threadIdx.x + st];
    __syncthreads();
  }
  if (threadIdx.x == 0) y[c] = s[0] / M;
}

__global__ void scale_kernel(float* __restrict__ f, const float* __restrict__ y, int N) {
  long i = (long)blockIdx.x * TPB + threadIdx.x;
  if (i >= (long)128 * N) return;
  f[i] *= y[i / N];
}

__global__ void lrelu_ip_kernel(float* __restrict__ f, long n) {
  long i = (long)blockIdx.x * TPB + threadIdx.x;
  if (i >= n) return;
  float v = f[i];
  f[i] = v > 0.f ? v : 0.2f * v;
}

// score (8 x N) -> log_softmax over 8, written (N x 8)
__global__ void lsm_kernel(const float* __restrict__ score, float* __restrict__ out, int N) {
  int n = blockIdx.x * TPB + threadIdx.x;
  if (n >= N) return;
  float s[8], mx = -INFINITY;
  for (int k = 0; k < 8; ++k) { s[k] = score[(size_t)k * N + n]; mx = fmaxf(mx, s[k]); }
  float se = 0.f;
  for (int k = 0; k < 8; ++k) se += expf(s[k] - mx);
  float l = mx + logf(se);
  for (int k = 0; k < 8; ++k) out[(size_t)n * 8 + k] = s[k] - l;
}

// ---------------------------------------------------------------------------
static inline long cdivl(long a, long b) { return (a + b - 1) / b; }

extern "C" void kernel_launch(void* const* d_in, const int* in_sizes, int n_in,
                              void* d_out, int out_size, void* d_ws, size_t ws_size,
                              hipStream_t stream) {
  const int N = NPTS;
  const long NN = (long)N * N;
  const float* xin  = (const float*)d_in[0];   // (24 x 8000)
  const int*   face = (const int*)d_in[1];     // (8000 x 3)
  auto P = [&](int i) { return (const float*)d_in[i]; };

  // ---- workspace bump allocator (deterministic each call) ----
  char* base = (char*)d_ws;
  size_t off = 0;
  auto alloc = [&](size_t bytes) -> void* {
    void* p = base + off;
    off += (bytes + 255) & ~(size_t)255;
    return p;
  };

  float* fPD    = (float*)alloc(NN * 4);                  // pd f32; later adj bf16
  bf16t* bADJ   = (bf16t*)fPD;
  bf16t* bFEATT = (bf16t*)alloc((size_t)N * 40 * 128 * 2);
  float* fYA    = (float*)alloc((size_t)N * 40 * 64 * 4);
  bf16t* bTT    = (bf16t*)alloc((size_t)N * 40 * 64 * 2);
  float* fCOOR0 = (float*)alloc((size_t)64 * N * 4);
  bf16t* bCOORT = (bf16t*)alloc((size_t)N * 64 * 2);
  float* fXX    = (float*)alloc((size_t)N * 4);
  int*   iIDX40 = (int*)alloc((size_t)N * 40 * 4);
  int*   iIDX20 = (int*)alloc((size_t)N * 20 * 4);
  float* fXK1   = (float*)alloc((size_t)64 * N * 4);
  float* fCOOR2 = (float*)alloc((size_t)64 * N * 4);
  bf16t* bC2T   = (bf16t*)alloc((size_t)N * 64 * 2);
  float* fCOORF = (float*)alloc((size_t)128 * N * 4);
  bf16t* bCMT   = (bf16t*)alloc((size_t)N * 128 * 2);
  float* fNX0   = (float*)alloc((size_t)512 * N * 4);
  float* fNX1   = (float*)alloc((size_t)512 * N * 4);
  float* fYT    = (float*)alloc((size_t)512 * N * 4);
  float* fHS    = (float*)alloc((size_t)512 * N * 4);
  float* fHA    = (float*)alloc((size_t)512 * N * 4);
  float* fHB    = (float*)alloc((size_t)512 * N * 4);
  float* fQB    = (float*)alloc((size_t)512 * N * 4);
  bf16t* bQCN   = (bf16t*)alloc((size_t)512 * N * 2);
  bf16t* bXT8K  = (bf16t*)alloc((size_t)N * 512 * 2);
  float* fX1A   = (float*)alloc((size_t)256 * N * 4);
  float* fX1B   = (float*)alloc((size_t)128 * N * 4);
  float* fATTN  = (float*)alloc((size_t)N * 4);
  float* fATTC  = (float*)alloc((size_t)N * 4);
  float* fFBUF  = (float*)alloc((size_t)128 * N * 4);
  float* fGBUF  = (float*)alloc((size_t)128 * N * 4);
  float* fSCORE = (float*)alloc((size_t)8 * N * 4);
  float* fYV0   = (float*)alloc(128 * 4);
  float* fYV1   = (float*)alloc(32 * 4);
  float* fYV2   = (float*)alloc(128 * 4);
  float* dMEAN  = (float*)alloc(512 * 4);
  float* dVAR   = (float*)alloc(512 * 4);

  // ---- launch helpers ----
  auto mm = [&](const bf16t* A, const bf16t* B, float* C, const float* bias,
                int O, int K, int M) {
    dim3 g((unsigned)cdivl(M, 256), (unsigned)(O / 16));
    wmma_mm_kernel<<<g, 128, 0, stream>>>(A, B, C, bias, K, M);
  };
  auto stats = [&](const float* Y, int C, int M) {
    stats_kernel<<<C, TPB, 0, stream>>>(Y, dMEAN, dVAR, M);
  };
  auto bnact = [&](const float* Y, int gi, int bi, const float* resid, float* out,
                   int C, int M) {
    long tot = (long)C * M;
    bn_act_kernel<<<(unsigned)cdivl(tot, TPB), TPB, 0, stream>>>(
        Y, P(gi), P(bi), dMEAN, dVAR, resid, out, M, tot);
  };
  auto tr = [&](const float* s, bf16t* d, int O, int M) {
    long tot = (long)O * M;
    tr_bf16_kernel<<<(unsigned)cdivl(tot, TPB), TPB, 0, stream>>>(s, d, O, M);
  };
  auto cvt = [&](const float* s, bf16t* d, long n) {
    f32_to_bf16_kernel<<<(unsigned)cdivl(n, TPB), TPB, 0, stream>>>(s, d, n);
  };
  auto smm = [&](const float* W, const float* X, const float* bias, float* out,
                 int O, int K, int M, int act) {
    long tot = (long)O * M;
    smm_kernel<<<(unsigned)cdivl(tot, TPB), TPB, 0, stream>>>(W, X, bias, out, O, K, M, act);
  };
  auto mkw = [&](int pi, int elems) -> bf16t* {
    bf16t* p = (bf16t*)alloc((size_t)elems * 2);
    cvt(P(pi), p, elems);
    return p;
  };

  // ---- bf16 weight copies (WMMA A operands) ----
  bf16t* w_c1c2 = mkw(5, 64 * 64);
  bf16t* w_m1 = mkw(6, 64 * 128);
  bf16t* w_m2 = mkw(9, 64 * 64);
  bf16t* w_m3 = mkw(12, 64 * 128);
  bf16t* w_m4 = mkw(15, 64 * 64);
  bf16t* w_m5 = mkw(18, 64 * 128);
  bf16t* w_c2c1 = mkw(21, 128 * 64);
  bf16t* w_c2c2 = mkw(24, 128 * 128);
  bf16t* w_c1n2 = mkw(28, 64 * 64);
  int cins[4] = {64, 128, 256, 512}, couts[4] = {128, 256, 512, 512};
  bf16t *w_head[4], *w_gw1[4][2], *w_gw2[4][2], *w_tail[4];
  for (int i = 0; i < 4; ++i) {
    int b0 = 29 + i * 18, cin = cins[i], cout = couts[i];
    w_head[i] = mkw(b0, cout * cin);
    for (int j = 0; j < 2; ++j) {
      int jb = b0 + 3 + j * 6;
      w_gw1[i][j] = mkw(jb, cout * cout);
      w_gw2[i][j] = mkw(jb + 4, cout * cout);
    }
    w_tail[i] = mkw(b0 + 15, cout * cout);
  }
  bf16t* w_r1 = mkw(101, 256 * 512);
  bf16t* w_r2 = mkw(104, 128 * 256);
  bf16t* w_p3 = mkw(113, 128 * 128);

  // ================= coor stem: c1c =================
  smm(P(2), xin, nullptr, fHA, 64, 12, N, 0);          // c1c_w1 (K=12)
  stats(fHA, 64, N); bnact(fHA, 3, 4, nullptr, fHA, 64, N);
  tr(fHA, bXT8K, 64, N);
  mm(w_c1c2, bXT8K, fCOOR0, nullptr, 64, 64, N);       // coor (64 x 8000)

  // ================= KNN (shared for k=40 and k=20) =================
  colsumsq_kernel<<<(unsigned)cdivl(N, TPB), TPB, 0, stream>>>(fCOOR0, fXX, 64, N);
  tr(fCOOR0, bCOORT, 64, N);
  mm(bCOORT, bCOORT, fPD, nullptr, N, 64, N);          // Gram 8000x8000 via WMMA
  pd_kernel<<<(unsigned)cdivl(NN, TPB), TPB, 0, stream>>>(fPD, fXX, N);
  topk_kernel<<<N, TPB, 0, stream>>>(fPD, iIDX40, N, 40);
  topk_kernel<<<N, TPB, 0, stream>>>(fPD, iIDX20, N, 20);
  // pd region now reused for adjacency (bf16)
  adj_gen_kernel<<<(unsigned)cdivl(NN, TPB), TPB, 0, stream>>>(face, bADJ, N);

  // ================= mgm =================
  long g40 = (long)N * 40 * 128;
  gather_feat_kernel<<<(unsigned)cdivl(g40, TPB), TPB, 0, stream>>>(fCOOR0, iIDX40, bFEATT, N, 40);
  mm(w_m1, bFEATT, fYA, nullptr, 64, 128, N * 40);
  stats(fYA, 64, N * 40); bnact(fYA, 7, 8, nullptr, fYA, 64, N * 40);
  tr(fYA, bTT, 64, N * 40);
  mm(w_m2, bTT, fYA, nullptr, 64, 64, N * 40);
  stats(fYA, 64, N * 40); bnact(fYA, 10, 11, nullptr, fYA, 64, N * 40);
  maxk_kernel<<<(unsigned)cdivl((long)64 * N, TPB), TPB, 0, stream>>>(fYA, fXK1, 64, N, 40);

  long g20 = (long)N * 20 * 128;
  gather_feat_kernel<<<(unsigned)cdivl(g20, TPB), TPB, 0, stream>>>(fCOOR0, iIDX20, bFEATT, N, 20);
  mm(w_m3, bFEATT, fYA, nullptr, 64, 128, N * 20);
  stats(fYA, 64, N * 20); bnact(fYA, 13, 14, nullptr, fYA, 64, N * 20);
  tr(fYA, bTT, 64, N * 20);
  mm(w_m4, bTT, fYA, nullptr, 64, 64, N * 20);
  stats(fYA, 64, N * 20); bnact(fYA, 16, 17, nullptr, fYA, 64, N * 20);   // f2n = fYA
  build_cat_kernel<<<(unsigned)cdivl(g20, TPB), TPB, 0, stream>>>(fYA, fXK1, bFEATT, N, 20);
  mm(w_m5, bFEATT, fYA, nullptr, 64, 128, N * 20);
  stats(fYA, 64, N * 20); bnact(fYA, 19, 20, nullptr, fYA, 64, N * 20);
  maxk_kernel<<<(unsigned)cdivl((long)64 * N, TPB), TPB, 0, stream>>>(fYA, fCOOR2, 64, N, 20);

  // ================= c2c =================
  tr(fCOOR2, bC2T, 64, N);
  mm(w_c2c1, bC2T, fHA, nullptr, 128, 64, N);
  stats(fHA, 128, N); bnact(fHA, 22, 23, nullptr, fHA, 128, N);
  tr(fHA, bCMT, 128, N);
  mm(w_c2c2, bCMT, fCOORF, nullptr, 128, 128, N);      // coor final (128 x 8000)

  // ================= nor stem: c1n =================
  smm(P(25), xin + (size_t)12 * N, nullptr, fHA, 64, 12, N, 0);
  stats(fHA, 64, N); bnact(fHA, 26, 27, nullptr, fHA, 64, N);
  tr(fHA, bXT8K, 64, N);
  mm(w_c1n2, bXT8K, fNX0, nullptr, 64, 64, N);

  // ================= GCN blocks =================
  float* xg = fNX0;
  float* xo = fNX1;
  for (int i = 0; i < 4; ++i) {
    int b0 = 29 + i * 18, cin = cins[i], cout = couts[i];
    tr(xg, bXT8K, cin, N);
    mm(w_head[i], bXT8K, fYT, nullptr, cout, cin, N);
    stats(fYT, cout, N); bnact(fYT, b0 + 1, b0 + 2, nullptr, fHS, cout, N);  // sc = fHS
    float* hcur = fHS;
    for (int j = 0; j < 2; ++j) {
      int jb = b0 + 3 + j * 6;
      tr(hcur, bXT8K, cout, N);
      mm(w_gw1[i][j], bXT8K, fYT, P(jb + 1), cout, cout, N);                 // + b1
      stats(fYT, cout, N); bnact(fYT, jb + 2, jb + 3, nullptr, fYT, cout, N);
      tr(fYT, bXT8K, cout, N);
      mm(w_gw2[i][j], bXT8K, fQB, P(jb + 5), cout, cout, N);                 // + b2
      cvt(fQB, bQCN, (long)cout * N);
      float* hn = (j == 0) ? fHA : fHB;
      mm(bQCN, bADJ, hn, nullptr, cout, N, N);   // adjacency matmul, K=8000 (L2-resident)
      hcur = hn;
    }
    tr(hcur, bXT8K, cout, N);
    mm(w_tail[i], bXT8K, fYT, nullptr, cout, cout, N);
    stats(fYT, cout, N); bnact(fYT, b0 + 16, b0 + 17, fHS, xo, cout, N);     // + sc
    float* t = xg; xg = xo; xo = t;
  }
  // xg = final nor (512 x 8000)

  // ================= r1 / r2 =================
  tr(xg, bXT8K, 512, N);
  mm(w_r1, bXT8K, fX1A, nullptr, 256, 512, N);
  stats(fX1A, 256, N); bnact(fX1A, 102, 103, nullptr, fX1A, 256, N);
  tr(fX1A, bXT8K, 256, N);
  mm(w_r2, bXT8K, fX1B, nullptr, 128, 256, N);
  stats(fX1B, 128, N); bnact(fX1B, 105, 106, nullptr, fX1B, 128, N);

  // ================= attention fusion + head =================
  smm(P(107), fCOORF, P(108), fATTC, 1, 128, N, 2);    // sigmoid(sac . coor)
  smm(P(109), fX1B,  P(110), fATTN, 1, 128, N, 2);     // sigmoid(san . x1)
  combine_kernel<<<(unsigned)cdivl((long)128 * N, TPB), TPB, 0, stream>>>(
      fCOORF, fX1B, fATTN, fATTC, fFBUF, N);
  meann_kernel<<<128, TPB, 0, stream>>>(fFBUF, fYV0, N);
  smm(P(111), fYV0, nullptr, fYV1, 32, 128, 1, 3);     // relu
  smm(P(112), fYV1, nullptr, fYV2, 128, 32, 1, 2);     // sigmoid
  scale_kernel<<<(unsigned)cdivl((long)128 * N, TPB), TPB, 0, stream>>>(fFBUF, fYV2, N);
  tr(fFBUF, bCMT, 128, N);
  mm(w_p3, bCMT, fGBUF, nullptr, 128, 128, N);
  lrelu_ip_kernel<<<(unsigned)cdivl((long)128 * N, TPB), TPB, 0, stream>>>(fGBUF, (long)128 * N);
  smm(P(114), fGBUF, nullptr, fSCORE, 8, 128, N, 0);
  lsm_kernel<<<(unsigned)cdivl(N, TPB), TPB, 0, stream>>>(fSCORE, (float*)d_out, N);

  (void)in_sizes; (void)n_in; (void)out_size; (void)ws_size;
}
